// Attention_33887291965577
// MI455X (gfx1250) — matrix operand order repaired and checked by reading, not compiled, not run on hardware
//
#include <hip/hip_runtime.h>
#include <hip/hip_bf16.h>

// ---------------------------------------------------------------------------
// MI455X (gfx1250) attention: bf16 WMMA everywhere, fp32 accumulation.
// All GEMM operands fit in the 192MB L2, so fragments are loaded straight
// from global (L2-resident) instead of LDS-staging; LDS is used only for the
// per-wave P-tile transpose in the flash-attention inner loop.
// ---------------------------------------------------------------------------

typedef __bf16 v16bf __attribute__((ext_vector_type(16)));
typedef __bf16 v8bf  __attribute__((ext_vector_type(8)));
typedef float  v8f   __attribute__((ext_vector_type(8)));
typedef float  v4f   __attribute__((ext_vector_type(4)));

#define WMMA_BF16(a, b, c) \
  __builtin_amdgcn_wmma_f32_16x16x32_bf16(false, (a), false, (b), (short)0, (c), false, false)

static constexpr int  Bb = 8;
static constexpr int  Nn = 1024;
static constexpr int  Cc = 1024;
static constexpr int  Hh = 16;
static constexpr int  Dd = 64;
static constexpr float ATT_SCALE = 0.125f;   // 1/sqrt(64)

// ---------------------------------------------------------------------------
// fp32 -> bf16 convert (grid-stride)
// ---------------------------------------------------------------------------
__global__ void cvt_f32_bf16(const float* __restrict__ src,
                             __bf16* __restrict__ dst, int n) {
  int i = blockIdx.x * blockDim.x + threadIdx.x;
  int stride = gridDim.x * blockDim.x;
  for (; i < n; i += stride) dst[i] = (__bf16)src[i];
}

// ---------------------------------------------------------------------------
// Fragment loaders (ISA 16-bit WMMA layouts, wave32).
// A-frag 16x32: lane m=L&15 is row M; K order: e<8 -> k0+half*8+e,
//               e>=8 -> k0+16+half*8+(e-8).
// B-frag 32x16: lane m=L&15 is col N; same K order as A-frag.
// ---------------------------------------------------------------------------
__device__ __forceinline__ v16bf load_a_frag(const __bf16* row, int k0, int half) {
  v8bf lo = *(const v8bf*)(row + k0 + half * 8);
  v8bf hi = *(const v8bf*)(row + k0 + 16 + half * 8);
  return __builtin_shufflevector(lo, hi, 0, 1, 2, 3, 4, 5, 6, 7,
                                         8, 9, 10, 11, 12, 13, 14, 15);
}
__device__ __forceinline__ v16bf load_b_frag(const __bf16* col, int k0, int half) {
  v8bf lo = *(const v8bf*)(col + k0 + half * 8);
  v8bf hi = *(const v8bf*)(col + k0 + 16 + half * 8);
  return __builtin_shufflevector(lo, hi, 0, 1, 2, 3, 4, 5, 6, 7,
                                         8, 9, 10, 11, 12, 13, 14, 15);
}

// ---------------------------------------------------------------------------
// Kernel 2: QKV GEMM.  out[gi,o] = sum_c x[gi,c]*w_qkv[o,c] + b_qkv[o]
// One 16x16 tile per wave; epilogue scatters into Q[B,H,N,D], K[B,H,N,D],
// and V transposed Vt[B,H,D,N] (so P@V B-fragments are contiguous later).
// ---------------------------------------------------------------------------
__global__ __launch_bounds__(256) void qkv_gemm(
    const __bf16* __restrict__ X, const __bf16* __restrict__ W,
    const float* __restrict__ bias,
    __bf16* __restrict__ qb, __bf16* __restrict__ kb, __bf16* __restrict__ vtb) {
  const int lane = threadIdx.x & 31;
  const int wave = threadIdx.x >> 5;
  const int mrow = lane & 15;
  const int half = lane >> 4;
  const int rowBase = blockIdx.x * 16;                 // over B*N = 8192 rows
  const int colBase = (blockIdx.y * 8 + wave) * 16;    // over 3C = 3072 cols

  const __bf16* arow = X + (size_t)(rowBase + mrow) * Cc;
  const __bf16* wcol = W + (size_t)(colBase + mrow) * Cc;

  v8f acc = {};
#pragma unroll 4
  for (int k0 = 0; k0 < Cc; k0 += 32) {
    __builtin_prefetch(arow + k0 + 256, 0, 0);   // global_prefetch_b8 into L2
    __builtin_prefetch(wcol + k0 + 256, 0, 0);
    v16bf af = load_a_frag(arow, k0, half);
    v16bf bf = load_b_frag(wcol, k0, half);
    acc = WMMA_BF16(af, bf, acc);
  }

  const int o   = colBase + mrow;
  const float bv = bias[o];
  const int s   = o >> 10;          // 0:q 1:k 2:v
  const int rem = o & 1023;
  const int h   = rem >> 6;
  const int d   = rem & 63;
#pragma unroll
  for (int r = 0; r < 8; ++r) {
    const int gi = rowBase + r + half * 8;
    const int b  = gi >> 10;
    const int n  = gi & 1023;
    const __bf16 val = (__bf16)(acc[r] + bv);
    const size_t bh = (size_t)b * Hh + h;
    if (s == 0)      qb[(bh * Nn + n) * Dd + d] = val;
    else if (s == 1) kb[(bh * Nn + n) * Dd + d] = val;
    else             vtb[(bh * Dd + d) * Nn + n] = val;
  }
}

// ---------------------------------------------------------------------------
// Kernel 3: flash attention. 1 wave = 16 queries x full D=64.
// Online softmax over 32-key chunks; 8 WMMAs per chunk (4 QK^T + 4 PV).
// P tile transposed D-layout -> A-layout through per-wave LDS (DS ops are
// in-order within a wave, so no barrier is needed for the private region).
// ---------------------------------------------------------------------------
__global__ __launch_bounds__(128) void flash_attn(
    const __bf16* __restrict__ Qb, const __bf16* __restrict__ Kb,
    const __bf16* __restrict__ Vtb, __bf16* __restrict__ ctx) {
  __shared__ float lds_p[4][16 * 32];

  const int lane = threadIdx.x & 31;
  const int wave = threadIdx.x >> 5;
  const int mrow = lane & 15;
  const int half = lane >> 4;

  const int gt = blockIdx.x * 4 + wave;   // 0 .. B*H*(N/16)-1 = 8191
  const int bh = gt >> 6;                 // b*H + h
  const int qt = gt & 63;                 // query tile
  const int b  = bh >> 4;
  const int h  = bh & 15;

  const __bf16* Q  = Qb  + (size_t)bh * Nn * Dd;
  const __bf16* K  = Kb  + (size_t)bh * Nn * Dd;
  const __bf16* Vt = Vtb + (size_t)bh * Dd * Nn;

  // Q A-fragments (whole head dim, two 32-wide K chunks), reused all loop.
  const __bf16* qrow = Q + (size_t)(qt * 16 + mrow) * Dd;
  const v16bf qa0 = load_a_frag(qrow, 0, half);
  const v16bf qa1 = load_a_frag(qrow, 32, half);

  float rm[8], rl[8];
  v8f ov[4] = {};
#pragma unroll
  for (int r = 0; r < 8; ++r) { rm[r] = -3.0e38f; rl[r] = 0.0f; }

  float* lp = lds_p[wave];

  for (int k0 = 0; k0 < Nn; k0 += 32) {
    // ---- S = Q @ K^T for 32 keys (two 16-key score tiles) ----
    const __bf16* kc0 = K + (size_t)(k0 + mrow) * Dd;
    const __bf16* kc1 = K + (size_t)(k0 + 16 + mrow) * Dd;
    v8f s0 = {}, s1 = {};
    s0 = WMMA_BF16(qa0, load_b_frag(kc0, 0, half), s0);
    s0 = WMMA_BF16(qa1, load_b_frag(kc0, 32, half), s0);
    s1 = WMMA_BF16(qa0, load_b_frag(kc1, 0, half), s1);
    s1 = WMMA_BF16(qa1, load_b_frag(kc1, 32, half), s1);

    // ---- online softmax ----
    float mx[8];
#pragma unroll
    for (int r = 0; r < 8; ++r) {
      s0[r] *= ATT_SCALE; s1[r] *= ATT_SCALE;
      mx[r] = fmaxf(s0[r], s1[r]);
    }
#pragma unroll
    for (int off = 1; off < 16; off <<= 1)
#pragma unroll
      for (int r = 0; r < 8; ++r)
        mx[r] = fmaxf(mx[r], __shfl_xor(mx[r], off, 32));

    float alpha[8], ps[8];
#pragma unroll
    for (int r = 0; r < 8; ++r) {
      const float mn = fmaxf(rm[r], mx[r]);
      alpha[r] = __expf(rm[r] - mn);
      rm[r] = mn;
      s0[r] = __expf(s0[r] - mn);
      s1[r] = __expf(s1[r] - mn);
      ps[r] = s0[r] + s1[r];
    }
#pragma unroll
    for (int off = 1; off < 16; off <<= 1)
#pragma unroll
      for (int r = 0; r < 8; ++r)
        ps[r] += __shfl_xor(ps[r], off, 32);
#pragma unroll
    for (int r = 0; r < 8; ++r) rl[r] = rl[r] * alpha[r] + ps[r];
#pragma unroll
    for (int t = 0; t < 4; ++t)
#pragma unroll
      for (int r = 0; r < 8; ++r) ov[t][r] *= alpha[r];

    // ---- transpose P (C/D layout -> A layout) via per-wave LDS ----
#pragma unroll
    for (int r = 0; r < 8; ++r) {
      const int row = r + half * 8;
      lp[row * 32 + mrow]      = s0[r];
      lp[row * 32 + 16 + mrow] = s1[r];
    }
    const float* ar = lp + mrow * 32 + half * 8;
    const v4f p0 = *(const v4f*)(ar);
    const v4f p1 = *(const v4f*)(ar + 4);
    const v4f p2 = *(const v4f*)(ar + 16);
    const v4f p3 = *(const v4f*)(ar + 20);
    v16bf pa;
#pragma unroll
    for (int e = 0; e < 4; ++e) {
      pa[e]      = (__bf16)p0[e];
      pa[e + 4]  = (__bf16)p1[e];
      pa[e + 8]  = (__bf16)p2[e];
      pa[e + 12] = (__bf16)p3[e];
    }

    // ---- O += P @ V  (V stored transposed: [d][n] -> contiguous B-frags) ----
#pragma unroll
    for (int t = 0; t < 4; ++t) {
      const __bf16* vc = Vt + (size_t)(t * 16 + mrow) * Nn;
      ov[t] = WMMA_BF16(pa, load_b_frag(vc, k0, half), ov[t]);
    }
  }

  // ---- normalize and write context [B,N,C] in bf16 ----
#pragma unroll
  for (int r = 0; r < 8; ++r) rl[r] = 1.0f / rl[r];
#pragma unroll
  for (int t = 0; t < 4; ++t)
#pragma unroll
    for (int r = 0; r < 8; ++r) {
      const int n = qt * 16 + r + half * 8;
      const int c = h * 64 + t * 16 + mrow;
      ctx[((size_t)b * Nn + n) * Cc + c] = (__bf16)(ov[t][r] * rl[r]);
    }
}

// ---------------------------------------------------------------------------
// Kernel 4: output projection. out[gi,o] = sum_c ctx[gi,c]*w_proj[o,c] + b[o]
// ---------------------------------------------------------------------------
__global__ __launch_bounds__(256) void proj_gemm(
    const __bf16* __restrict__ Xc, const __bf16* __restrict__ W,
    const float* __restrict__ bias, float* __restrict__ out) {
  const int lane = threadIdx.x & 31;
  const int wave = threadIdx.x >> 5;
  const int mrow = lane & 15;
  const int half = lane >> 4;
  const int rowBase = blockIdx.x * 16;
  const int colBase = (blockIdx.y * 8 + wave) * 16;

  const __bf16* arow = Xc + (size_t)(rowBase + mrow) * Cc;
  const __bf16* wcol = W + (size_t)(colBase + mrow) * Cc;

  v8f acc = {};
#pragma unroll 4
  for (int k0 = 0; k0 < Cc; k0 += 32) {
    __builtin_prefetch(arow + k0 + 256, 0, 0);
    __builtin_prefetch(wcol + k0 + 256, 0, 0);
    v16bf af = load_a_frag(arow, k0, half);
    v16bf bf = load_b_frag(wcol, k0, half);
    acc = WMMA_BF16(af, bf, acc);
  }

  const int o = colBase + mrow;
  const float bv = bias[o];
#pragma unroll
  for (int r = 0; r < 8; ++r) {
    const int gi = rowBase + r + half * 8;
    out[(size_t)gi * Cc + o] = acc[r] + bv;
  }
}

// ---------------------------------------------------------------------------
// Host launch
// ---------------------------------------------------------------------------
extern "C" void kernel_launch(void* const* d_in, const int* in_sizes, int n_in,
                              void* d_out, int out_size, void* d_ws, size_t ws_size,
                              hipStream_t stream) {
  (void)in_sizes; (void)n_in; (void)out_size; (void)ws_size;
  const float* x      = (const float*)d_in[0];
  const float* w_qkv  = (const float*)d_in[1];
  const float* b_qkv  = (const float*)d_in[2];
  const float* w_proj = (const float*)d_in[3];
  const float* b_proj = (const float*)d_in[4];
  float* out = (float*)d_out;

  char* ws = (char*)d_ws;
  size_t off = 0;
  __bf16* x_bf     = (__bf16*)(ws + off); off += (size_t)Bb * Nn * Cc * 2;   // 16 MB
  __bf16* wqkv_bf  = (__bf16*)(ws + off); off += (size_t)3 * Cc * Cc * 2;    //  6 MB
  __bf16* wproj_bf = (__bf16*)(ws + off); off += (size_t)Cc * Cc * 2;        //  2 MB
  __bf16* q_bf     = (__bf16*)(ws + off); off += (size_t)Bb * Hh * Nn * Dd * 2; // 16 MB
  __bf16* k_bf     = (__bf16*)(ws + off); off += (size_t)Bb * Hh * Nn * Dd * 2; // 16 MB
  __bf16* vt_bf    = (__bf16*)(ws + off); off += (size_t)Bb * Hh * Dd * Nn * 2; // 16 MB
  __bf16* ctx_bf   = (__bf16*)(ws + off); off += (size_t)Bb * Nn * Cc * 2;   // 16 MB

  cvt_f32_bf16<<<2048, 256, 0, stream>>>(x, x_bf, Bb * Nn * Cc);
  cvt_f32_bf16<<<2048, 256, 0, stream>>>(w_qkv, wqkv_bf, 3 * Cc * Cc);
  cvt_f32_bf16<<<1024, 256, 0, stream>>>(w_proj, wproj_bf, Cc * Cc);

  // 8192 rows (B*N) / 16, 3072 cols / (8 waves * 16)
  qkv_gemm<<<dim3(512, 24), 256, 0, stream>>>(x_bf, wqkv_bf, b_qkv,
                                              q_bf, k_bf, vt_bf);
  // B*H*(N/16) = 8192 wave-tiles, 4 waves/block
  flash_attn<<<2048, 128, 0, stream>>>(q_bf, k_bf, vt_bf, ctx_bf);
  // 8192 rows / 16, 1024 cols / 128
  proj_gemm<<<dim3(512, 8), 256, 0, stream>>>(ctx_bf, wproj_bf, b_proj, out);
}
